// IMULSTMEncoder_14259291423009
// MI455X (gfx1250) — matrix operand order repaired and checked by reading, not compiled, not run on hardware
//
#include <hip/hip_runtime.h>
#include <hip/hip_bf16.h>

// ---------------------------------------------------------------------------
// IMU bidirectional-LSTM encoder for MI455X (gfx1250, wave32, WMMA).
//   swz_kernel        : weights fp32 -> f16 in WMMA-B fragment order
//   inproj_kernel     : [N*L,6] @ w_in^T + b, ReLU -> f16 x0
//   gates_gemm_kernel : time-major [16 seq x 512] tiles of x @ w_ih^T + bias,
//                       stored in C-fragment order (1 b128 per lane per tile)
//   lstm_dir_kernel   : 128-step recurrence; w_hh resident in LDS, h in LDS as
//                       A-operand tile, gate fragments loaded as single b128
//   head_kernel       : hidden @ w_out^T + b, LayerNorm, ReLU, len>0 mask
// ---------------------------------------------------------------------------

typedef __attribute__((ext_vector_type(16))) _Float16 v16h;
typedef __attribute__((ext_vector_type(8)))  _Float16 v8h;
typedef __attribute__((ext_vector_type(8)))  float    v8f;

#define N_SEQ     1024      // B*T
#define SEQ_L     128       // L
#define HDIM      128       // H
#define GDIM      512       // 4*H
#define PROJ_DIM  64
#define IN_DIM_C  6

__device__ __forceinline__ float sigmoidf_(float x) {
  return 1.0f / (1.0f + __expf(-x));
}

// ---- A-matrix fragment (16x32 f16) from row-major f16 buffer, rows m ----
// ISA layout: lanes 0-15 hold row m, K {0..7,16..23}; lanes 16-31 K {8..15,24..31}
__device__ __forceinline__ v16h load_a_frag_g(const _Float16* x, int row_stride,
                                              size_t mbase, int kbase) {
  const int lane = threadIdx.x & 31;
  const size_t m = mbase + (lane & 15);
  const int k0 = kbase + ((lane & 16) ? 8 : 0);
  const _Float16* p = x + m * (size_t)row_stride + k0;
  v8h lo = *(const v8h*)p;
  v8h hi = *(const v8h*)(p + 16);
  v16h r;
#pragma unroll
  for (int j = 0; j < 8; ++j) { r[j] = lo[j]; r[j + 8] = hi[j]; }
  return r;
}

__device__ __forceinline__ v16h load_a_frag_lds(const _Float16* hbuf, int hstride,
                                                int mbase, int kbase) {
  const int lane = threadIdx.x & 31;
  const int m = mbase + (lane & 15);
  const int k0 = kbase + ((lane & 16) ? 8 : 0);
  const _Float16* p = hbuf + m * hstride + k0;
  v8h lo = *(const v8h*)p;
  v8h hi = *(const v8h*)(p + 16);
  v16h r;
#pragma unroll
  for (int j = 0; j < 8; ++j) { r[j] = lo[j]; r[j + 8] = hi[j]; }
  return r;
}

// ---------------------------------------------------------------------------
// Weight fragment swizzle: W [COLS,K] row-major f32 -> f16 B-fragments.
// Fragment (kt,ct): 512 halves; lane L (col = ct*16+(L&15)) holds
// B[kt*32 + (L<16?0:8) + (j<8?j:j+8)][col], j = 0..15 (one 32B chunk/lane).
// ---------------------------------------------------------------------------
__global__ void swz_kernel(const float* __restrict__ W, _Float16* __restrict__ dst,
                           int K, int COLS) {
  const int total = K * COLS;
  const int nct = COLS >> 4;
  for (int idx = blockIdx.x * blockDim.x + threadIdx.x; idx < total;
       idx += gridDim.x * blockDim.x) {
    const int frag = idx >> 9;
    const int rem  = idx & 511;
    const int lane = rem >> 4;
    const int j    = rem & 15;
    const int kt = frag / nct;
    const int ct = frag % nct;
    const int col = ct * 16 + (lane & 15);
    const int k   = kt * 32 + ((lane & 16) ? 8 : 0) + (j < 8 ? j : j + 8);
    dst[idx] = (_Float16)W[(size_t)col * K + k];
  }
}

// ---------------------------------------------------------------------------
// Input projection: relu(imu @ w_in^T + b_in), 6 -> 64, f16 (sequence-major).
// ---------------------------------------------------------------------------
__global__ void inproj_kernel(const float* __restrict__ imu,
                              const float* __restrict__ w_in,
                              const float* __restrict__ b_in,
                              _Float16* __restrict__ x0) {
  const size_t idx = (size_t)blockIdx.x * blockDim.x + threadIdx.x;
  const int p = (int)(idx & 63);
  const size_t row = idx >> 6;
  const float* xi = imu + row * IN_DIM_C;
  float acc = b_in[p];
#pragma unroll
  for (int d = 0; d < IN_DIM_C; ++d) acc += xi[d] * w_in[p * IN_DIM_C + d];
  x0[idx] = (_Float16)fmaxf(acc, 0.f);
}

// ---------------------------------------------------------------------------
// Input-gate GEMM, time-major tiling: wave gw -> (t = gw>>6, ntile = gw&63)
// computes rows n = ntile*16..+15 at timestep t, all 512 gate columns.
// Output stored in C-fragment order:
//   gates[((ntile*32 + ct)*128 + t)*256 + lane*8 .. +7]  (f16, 16B per lane)
// so the recurrent kernel initializes each accumulator with ONE b128 load.
// ---------------------------------------------------------------------------
template <int DIN>
__global__ void gates_gemm_kernel(const _Float16* __restrict__ x,
                                  const _Float16* __restrict__ wswz,
                                  const float* __restrict__ b_ih,
                                  const float* __restrict__ b_hh,
                                  _Float16* __restrict__ gates) {
  constexpr int NKT = DIN / 32;
  const int lane = threadIdx.x & 31;
  const int gw = blockIdx.x * (blockDim.x >> 5) + (threadIdx.x >> 5);
  const int t = gw >> 6;         // timestep
  const int ntile = gw & 63;     // 16-sequence tile
  const int coll = lane & 15;

  // A fragments: rows are sequences ntile*16+(lane&15), x is sequence-major
  const size_t rowoff = ((size_t)(ntile * 16 + coll) * SEQ_L + t) * DIN;
  v16h afr[NKT];
#pragma unroll
  for (int kt = 0; kt < NKT; ++kt) {
    const int k0 = kt * 32 + ((lane & 16) ? 8 : 0);
    v8h lo = *(const v8h*)(x + rowoff + k0);
    v8h hi = *(const v8h*)(x + rowoff + k0 + 16);
#pragma unroll
    for (int j = 0; j < 8; ++j) { afr[kt][j] = lo[j]; afr[kt][j + 8] = hi[j]; }
  }

  for (int ct = 0; ct < 32; ++ct) {
    const int col = ct * 16 + coll;
    const float bias = b_ih[col] + b_hh[col];
    v8f c;
#pragma unroll
    for (int r = 0; r < 8; ++r) c[r] = bias;
#pragma unroll
    for (int kt = 0; kt < NKT; ++kt) {
      v16h b = *(const v16h*)(wswz + ((size_t)(kt * 32 + ct) << 9) + lane * 16);
      c = __builtin_amdgcn_wmma_f32_16x16x32_f16(false, afr[kt], false, b,
                                                 (short)0, c, false, false);
    }
    v8h hh;
#pragma unroll
    for (int r = 0; r < 8; ++r) hh[r] = (_Float16)c[r];
    *(v8h*)(gates + (((size_t)(ntile * 32 + ct)) * SEQ_L + t) * 256 + lane * 8) = hh;
  }
}

// ---------------------------------------------------------------------------
// Recurrent kernel: one WG owns 32 sequences, loops all 128 timesteps.
// w_hh fragments live in LDS (copied once) so nothing is hoisted/spilled.
//   phase 1: c = gates_frag(b128) ; c += h_prev @ w_hh^T  (WMMA, A/B from LDS)
//   phase 2: LSTM cell elementwise + length masking; h -> LDS + global.
// ---------------------------------------------------------------------------
#define K3_ROWS 32
#define K3_GST  520                       // padded f32 gate stride
#define K3_HST  136                       // padded f16 h stride (16B aligned)
#define K3_GBYTES (K3_ROWS * K3_GST * 4)  // 66560
#define K3_HBYTES (K3_ROWS * K3_HST * 2)  // 8704
#define K3_WBYTES (GDIM * HDIM * 2)       // 131072
#define K3_SMEM   (K3_GBYTES + K3_HBYTES + K3_WBYTES)

__global__ void lstm_dir_kernel(const _Float16* __restrict__ gates,
                                const _Float16* __restrict__ whh_swz,
                                const int* __restrict__ lengths,
                                _Float16* __restrict__ xnext,
                                _Float16* __restrict__ hid,
                                int backward, int dirOff, int writeHid) {
  extern __shared__ char smem_raw[];
  float*    gls = (float*)smem_raw;                              // [32][520]
  _Float16* hls = (_Float16*)(smem_raw + K3_GBYTES);             // [32][136]
  _Float16* wls = (_Float16*)(smem_raw + K3_GBYTES + K3_HBYTES); // 64K halves

  const int tid  = threadIdx.x;
  const int lane = tid & 31;
  const int wv   = tid >> 5;                 // wave 0..7
  const int seqbase = blockIdx.x * K3_ROWS;
  const int coll = lane & 15;

  // zero h, copy swizzled w_hh into LDS (16B chunks)
  for (int i = tid; i < K3_ROWS * K3_HST; i += blockDim.x) hls[i] = (_Float16)0.f;
  for (int i = tid * 8; i < GDIM * HDIM; i += blockDim.x * 8)
    *(v8h*)(wls + i) = *(const v8h*)(whh_swz + i);

  // elementwise ownership: col = tid&127, rows rgrp*16 + j
  const int ecol = tid & 127;
  const int rgrp = tid >> 7;
  float creg[16];
  int lenreg[16];
#pragma unroll
  for (int j = 0; j < 16; ++j) {
    creg[j] = 0.f;
    lenreg[j] = lengths[seqbase + rgrp * 16 + j];
  }
  __syncthreads();

  for (int step = 0; step < SEQ_L; ++step) {
    const int t = backward ? (SEQ_L - 1 - step) : step;

    // ---- phase 1: 8 output tiles (16x16) per wave, K = 128 ----
#pragma unroll
    for (int i = 0; i < 8; ++i) {
      const int tIdx = wv * 8 + i;
      const int rt = tIdx & 1;
      const int ct = tIdx >> 1;
      // accumulator init: one b128 of f16 gate values in C-fragment order
      const size_t gbase =
          (((size_t)((seqbase >> 4) + rt) * 32 + ct) * SEQ_L + t) * 256 + lane * 8;
      v8h g8 = *(const v8h*)(gates + gbase);
      v8f c;
#pragma unroll
      for (int r = 0; r < 8; ++r) c[r] = (float)g8[r];
#pragma unroll
      for (int kt = 0; kt < 4; ++kt) {
        v16h a = load_a_frag_lds(hls, K3_HST, rt * 16, kt * 32);
        v16h b = *(const v16h*)(wls + ((kt * 32 + ct) << 9) + lane * 16);
        c = __builtin_amdgcn_wmma_f32_16x16x32_f16(false, a, false, b,
                                                   (short)0, c, false, false);
      }
      const int rlo = (lane & 16) ? 8 : 0;
      const int col = ct * 16 + coll;
#pragma unroll
      for (int r = 0; r < 8; ++r)
        gls[(rt * 16 + r + rlo) * K3_GST + col] = c[r];
    }
    __syncthreads();

    // ---- phase 2: LSTM cell (gate order i,f,g,o) + masking ----
#pragma unroll
    for (int j = 0; j < 16; ++j) {
      const int r = rgrp * 16 + j;
      const float gi = gls[r * K3_GST + ecol];
      const float gf = gls[r * K3_GST + ecol + HDIM];
      const float gg = gls[r * K3_GST + ecol + 2 * HDIM];
      const float go = gls[r * K3_GST + ecol + 3 * HDIM];
      const float cn = sigmoidf_(gf) * creg[j] + sigmoidf_(gi) * tanhf(gg);
      const float hn = sigmoidf_(go) * tanhf(cn);
      float h;
      if (t < lenreg[j]) { creg[j] = cn; h = hn; }
      else               { h = (float)hls[r * K3_HST + ecol]; }   // hold
      hls[r * K3_HST + ecol] = (_Float16)h;
      xnext[(((size_t)(seqbase + r)) * SEQ_L + t) * 256 + dirOff + ecol] = (_Float16)h;
    }
    __syncthreads();
  }

  if (writeHid) {
#pragma unroll
    for (int j = 0; j < 16; ++j) {
      const int r = rgrp * 16 + j;
      hid[(size_t)(seqbase + r) * 256 + dirOff + ecol] = hls[r * K3_HST + ecol];
    }
  }
}

// ---------------------------------------------------------------------------
// Output head: y = hidden @ w_out^T + b_out; LayerNorm(256); ReLU; len>0 mask.
// LayerNorm is done in-register with 16-lane shuffle reductions (rows r and
// r+8 occupy disjoint lane halves of each C/D fragment).
// ---------------------------------------------------------------------------
__global__ void head_kernel(const _Float16* __restrict__ hid,
                            const _Float16* __restrict__ wout_swz,
                            const float* __restrict__ b_out,
                            const float* __restrict__ gamma,
                            const float* __restrict__ beta,
                            const int* __restrict__ lengths,
                            float* __restrict__ out) {
  const int lane = threadIdx.x & 31;
  const int gw = blockIdx.x * (blockDim.x >> 5) + (threadIdx.x >> 5);
  const size_t rowbase = (size_t)gw * 16;
  const int coll = lane & 15;
  const int rlo = (lane & 16) ? 8 : 0;

  v16h afr[8];
#pragma unroll
  for (int kt = 0; kt < 8; ++kt)
    afr[kt] = load_a_frag_g(hid, 256, rowbase, kt * 32);

  v8f acc[16];
  for (int ct = 0; ct < 16; ++ct) {
    const int col = ct * 16 + coll;
    v8f c;
#pragma unroll
    for (int r = 0; r < 8; ++r) c[r] = b_out[col];
#pragma unroll
    for (int kt = 0; kt < 8; ++kt) {
      v16h b = *(const v16h*)(wout_swz + ((size_t)(kt * 16 + ct) << 9) + lane * 16);
      c = __builtin_amdgcn_wmma_f32_16x16x32_f16(false, afr[kt], false, b,
                                                 (short)0, c, false, false);
    }
    acc[ct] = c;
  }

#pragma unroll
  for (int r = 0; r < 8; ++r) {
    float s = 0.f, s2 = 0.f;
#pragma unroll
    for (int ct = 0; ct < 16; ++ct) { float v = acc[ct][r]; s += v; s2 += v * v; }
#pragma unroll
    for (int m = 1; m < 16; m <<= 1) {
      s  += __shfl_xor(s, m);
      s2 += __shfl_xor(s2, m);
    }
    const float mu   = s * (1.f / 256.f);
    const float var  = s2 * (1.f / 256.f) - mu * mu;
    const float rstd = rsqrtf(var + 1e-5f);
    const size_t row = rowbase + r + rlo;
    const int keep = lengths[row] > 0;
#pragma unroll
    for (int ct = 0; ct < 16; ++ct) {
      const int col = ct * 16 + coll;
      float v = (acc[ct][r] - mu) * rstd * gamma[col] + beta[col];
      v = fmaxf(v, 0.f);
      out[row * 256 + col] = keep ? v : 0.f;
    }
  }
}

// ---------------------------------------------------------------------------
// Host launcher.  Input order assumes jax.tree_util leaves (dict keys sorted):
//   0 imu, 1 lengths, 2 b_in, 3 b_out, 4 beta, 5 gamma,
//   per layer l (base 6+8l): dir 'b' {b_hh,b_ih,w_hh,w_ih} then dir 'f' {...},
//   30 w_in, 31 w_out.
// ---------------------------------------------------------------------------
extern "C" void kernel_launch(void* const* d_in, const int* in_sizes, int n_in,
                              void* d_out, int out_size, void* d_ws, size_t ws_size,
                              hipStream_t stream) {
  (void)in_sizes; (void)n_in; (void)out_size; (void)ws_size;
  const float* imu     = (const float*)d_in[0];
  const int*   lengths = (const int*)d_in[1];
  const float* b_in    = (const float*)d_in[2];
  const float* b_out   = (const float*)d_in[3];
  const float* beta    = (const float*)d_in[4];
  const float* gamma   = (const float*)d_in[5];
  const float* w_in    = (const float*)d_in[30];
  const float* w_out   = (const float*)d_in[31];

  const float *Wih[3][2], *Whh[3][2], *Bih[3][2], *Bhh[3][2];
  for (int l = 0; l < 3; ++l) {
    const int base = 6 + 8 * l;
    // dir 0 = forward ('f'), dir 1 = backward ('b'); dict order is 'b' first.
    Bhh[l][1] = (const float*)d_in[base + 0];
    Bih[l][1] = (const float*)d_in[base + 1];
    Whh[l][1] = (const float*)d_in[base + 2];
    Wih[l][1] = (const float*)d_in[base + 3];
    Bhh[l][0] = (const float*)d_in[base + 4];
    Bih[l][0] = (const float*)d_in[base + 5];
    Whh[l][0] = (const float*)d_in[base + 6];
    Wih[l][0] = (const float*)d_in[base + 7];
  }

  // ---- workspace layout ----
  char* ws = (char*)d_ws;
  _Float16* wpool = (_Float16*)ws;
  size_t hoff = 0;
  _Float16 *wih_swz[3][2], *whh_swz[3][2];
  for (int l = 0; l < 3; ++l)
    for (int d = 0; d < 2; ++d) {
      wih_swz[l][d] = wpool + hoff; hoff += (size_t)GDIM * (l == 0 ? PROJ_DIM : 256);
      whh_swz[l][d] = wpool + hoff; hoff += (size_t)GDIM * HDIM;
    }
  _Float16* wout_swz = wpool + hoff; hoff += 256 * 256;

  size_t byteoff = (hoff * 2 + 255) & ~(size_t)255;
  _Float16* x0  = (_Float16*)(ws + byteoff); byteoff += (size_t)N_SEQ * SEQ_L * PROJ_DIM * 2;
  _Float16* xA  = (_Float16*)(ws + byteoff); byteoff += (size_t)N_SEQ * SEQ_L * 256 * 2;
  _Float16* xB  = (_Float16*)(ws + byteoff); byteoff += (size_t)N_SEQ * SEQ_L * 256 * 2;
  _Float16* gts = (_Float16*)(ws + byteoff); byteoff += (size_t)N_SEQ * SEQ_L * GDIM * 2;
  _Float16* hid = (_Float16*)(ws + byteoff); byteoff += (size_t)N_SEQ * 256 * 2;

  // ---- 1. weight conversion to WMMA fragment layout ----
  auto launch_swz = [&](const float* W, _Float16* dst, int K, int COLS) {
    const int total = K * COLS;
    swz_kernel<<<(total + 255) / 256, 256, 0, stream>>>(W, dst, K, COLS);
  };
  for (int l = 0; l < 3; ++l)
    for (int d = 0; d < 2; ++d) {
      launch_swz(Wih[l][d], wih_swz[l][d], l == 0 ? PROJ_DIM : 256, GDIM);
      launch_swz(Whh[l][d], whh_swz[l][d], HDIM, GDIM);
    }
  launch_swz(w_out, wout_swz, 256, 256);

  // ---- 2. input projection ----
  inproj_kernel<<<(N_SEQ * SEQ_L * PROJ_DIM) / 256, 256, 0, stream>>>(imu, w_in, b_in, x0);

  // ---- 3. stacked bidirectional LSTM layers ----
  const _Float16* xin = x0;
  _Float16* bufs[2] = { xA, xB };
  for (int l = 0; l < 3; ++l) {
    _Float16* xout = bufs[l & 1];
    for (int d = 0; d < 2; ++d) {
      if (l == 0)
        gates_gemm_kernel<PROJ_DIM><<<(SEQ_L * (N_SEQ / 16)) / 8, 256, 0, stream>>>(
            xin, wih_swz[l][d], Bih[l][d], Bhh[l][d], gts);
      else
        gates_gemm_kernel<256><<<(SEQ_L * (N_SEQ / 16)) / 8, 256, 0, stream>>>(
            xin, wih_swz[l][d], Bih[l][d], Bhh[l][d], gts);
      lstm_dir_kernel<<<N_SEQ / K3_ROWS, 256, K3_SMEM, stream>>>(
          gts, whh_swz[l][d], lengths, xout, hid,
          /*backward=*/d, /*dirOff=*/d * HDIM, /*writeHid=*/(l == 2) ? 1 : 0);
    }
    xin = xout;
  }

  // ---- 4. output head ----
  head_kernel<<<(N_SEQ / 16) / 8, 256, 0, stream>>>(hid, wout_swz, b_out, gamma, beta,
                                                    lengths, (float*)d_out);
}
